// Encoder_obs_model_1949915152527
// MI455X (gfx1250) — matrix operand order repaired and checked by reading, not compiled, not run on hardware
//
#include <hip/hip_runtime.h>
#include <hip/hip_bf16.h>

// ---------------------------------------------------------------------------
// ConvLSTM encoder for MI455X (gfx1250, wave32, WMMA).
//   k_conv1 : 1->4 5x5 conv + relu + 2x2 maxpool (VALU, fused)   -> x1h (f16)
//   k_conv2 : 4->4 5x5 conv + relu + 2x2 maxpool (VALU, fused)   -> x2h (f16)
//   k_convx : 4->32 5x5 conv via WMMA implicit GEMM (+bias)      -> xg  (f16)
//   k_lstm  : per-step 8->32 conv via WMMA + gate math, h/c in d_out
//
// Implicit-GEMM K ordering: k = cin*32 + tap (25 taps zero-padded to 32), so
// cin == kstep at compile time and each A element's LDS offset is a cndmask
// between two compile-time constants. Input tiles are staged zero-padded in
// LDS (no divergent global loads); weight panels are staged transposed [n][k]
// so B fragments load as 2x ds_load_b128.
// ---------------------------------------------------------------------------

typedef __attribute__((ext_vector_type(16))) _Float16 v16h;
typedef __attribute__((ext_vector_type(8)))  float    v8f;

#define TT   8
#define BB   32
#define HH   256
#define WW   256
#define H1   128
#define W1   128
#define HC   64
#define WC   64

#define TW   68              // padded tile row width (64 + 2*2 halo)
#define TR   8               // tile rows (6 real + 2 zero rows for pad taps)

__device__ inline float sigmoid_f(float x) {
  return 1.0f / (1.0f + __expf(-x));
}

__device__ inline float tanh_f(float x) {
#if __has_builtin(__builtin_amdgcn_tanhf)
  return __builtin_amdgcn_tanhf(x);
#else
  x = fminf(fmaxf(x, -15.0f), 15.0f);
  float e = __expf(2.0f * x);
  return (e - 1.0f) / (e + 1.0f);
#endif
}

// ---------------------------------------------------------------------------
// A fragment from the zero-padded LDS tile sA[cin][TR][TW].
// 16-bit A 16x32 layout (contiguous, matches B): element e -> K = 32*ks + e + 16*hi.
// With k = cin*32 + tap, cin == ks (compile time)
// and tap has exactly two compile-time candidates -> offset cndmask.
//   yloc68 = (y - y0) * TW,  xt = x0 + (lane&15)  (tile col, halo-shifted)
// ---------------------------------------------------------------------------
template <int KS>
__device__ inline v16h load_A_tile(const _Float16* __restrict__ sA,
                                   int yloc68, int xt, int hi) {
  v16h a;
  const _Float16* base = sA + KS * (TR * TW) + yloc68 + xt;
#pragma unroll
  for (int e = 0; e < 16; ++e) {
    const int tc   = e;
    const int off0 = (tc / 5) * TW + (tc % 5);            // hi = 0
    const int off1 = ((tc + 16) / 5) * TW + ((tc + 16) % 5); // hi = 1
    a[e] = base[hi ? off1 : off0];
  }
  return a;
}

// B fragment from the transposed LDS weight panel swpT[n][k], row stride KSR
// halves (KSR = Kpad + 16, keeps 32-byte alignment and staggers banks).
// Per-lane elements are 16 contiguous halves -> 2x ds_load_b128.
__device__ inline v16h load_B_panel(const _Float16* __restrict__ swpT,
                                    int KSR, int kstep, int ntile, int lane) {
  const int n  = (lane & 15) + ntile * 16;
  const int hi = lane >> 4;
  const _Float16* p = swpT + n * KSR + kstep * 32 + hi * 16;
  return *(const v16h*)p;
}

// ---------------------------------------------------------------------------
// k_conv1: obs (img,1,256,256) f32 -> x1h (img,4,128,128) f16, fused relu+pool
// ---------------------------------------------------------------------------
__global__ void k_conv1(const float* __restrict__ obs,
                        const float* __restrict__ w1,
                        const float* __restrict__ b1,
                        _Float16* __restrict__ x1h) {
  __shared__ float sw[100];
  __shared__ float sb[4];
  const int tid = threadIdx.x;
  if (tid < 100) sw[tid] = w1[tid];
  if (tid < 4)   sb[tid] = b1[tid];
  __syncthreads();

  const int gid = blockIdx.x * 256 + tid;       // 256 imgs * 128*128
  const int img = gid >> 14;
  const int rem = gid & 16383;
  const int yo  = rem >> 7;
  const int xo  = rem & 127;

  const float* src = obs + (size_t)img * HH * WW;
  const int yb = 2 * yo - 2, xb = 2 * xo - 2;

  float patch[6][6];
#pragma unroll
  for (int i = 0; i < 6; ++i)
#pragma unroll
    for (int j = 0; j < 6; ++j) {
      int yy = yb + i, xx = xb + j;
      patch[i][j] = (yy >= 0 && yy < HH && xx >= 0 && xx < WW)
                        ? src[yy * WW + xx] : 0.0f;
    }

#pragma unroll
  for (int co = 0; co < 4; ++co) {
    float a00 = sb[co], a01 = sb[co], a10 = sb[co], a11 = sb[co];
#pragma unroll
    for (int ky = 0; ky < 5; ++ky)
#pragma unroll
      for (int kx = 0; kx < 5; ++kx) {
        float wv = sw[co * 25 + ky * 5 + kx];
        a00 = fmaf(wv, patch[ky][kx],         a00);
        a01 = fmaf(wv, patch[ky][kx + 1],     a01);
        a10 = fmaf(wv, patch[ky + 1][kx],     a10);
        a11 = fmaf(wv, patch[ky + 1][kx + 1], a11);
      }
    float r = fmaxf(fmaxf(a00, a01), fmaxf(a10, a11));
    r = fmaxf(r, 0.0f);   // relu commutes with max
    x1h[(((size_t)img * 4 + co) * H1 + yo) * W1 + xo] = (_Float16)r;
  }
}

// ---------------------------------------------------------------------------
// k_conv2: x1h (img,4,128,128) f16 -> x2h (img,4,64,64) f16, fused relu+pool
// ---------------------------------------------------------------------------
__global__ void k_conv2(const _Float16* __restrict__ x1h,
                        const float* __restrict__ w2,
                        const float* __restrict__ b2,
                        _Float16* __restrict__ x2h) {
  __shared__ float sw[400];
  __shared__ float sb[4];
  const int tid = threadIdx.x;
  for (int i = tid; i < 400; i += 256) sw[i] = w2[i];
  if (tid < 4) sb[tid] = b2[tid];
  __syncthreads();

  const int gid = blockIdx.x * 256 + tid;       // 256 imgs * 64*64
  const int img = gid >> 12;
  const int rem = gid & 4095;
  const int yo  = rem >> 6;
  const int xo  = rem & 63;

  const _Float16* src = x1h + (size_t)img * 4 * H1 * W1;
  const int yb = 2 * yo - 2, xb = 2 * xo - 2;

  float acc[4][4];
#pragma unroll
  for (int co = 0; co < 4; ++co)
#pragma unroll
    for (int p = 0; p < 4; ++p) acc[co][p] = sb[co];

  for (int cin = 0; cin < 4; ++cin) {
    float patch[6][6];
    const _Float16* sp = src + (size_t)cin * H1 * W1;
#pragma unroll
    for (int i = 0; i < 6; ++i)
#pragma unroll
      for (int j = 0; j < 6; ++j) {
        int yy = yb + i, xx = xb + j;
        patch[i][j] = (yy >= 0 && yy < H1 && xx >= 0 && xx < W1)
                          ? (float)sp[yy * W1 + xx] : 0.0f;
      }
#pragma unroll
    for (int co = 0; co < 4; ++co)
#pragma unroll
      for (int ky = 0; ky < 5; ++ky)
#pragma unroll
        for (int kx = 0; kx < 5; ++kx) {
          float wv = sw[(co * 4 + cin) * 25 + ky * 5 + kx];
          acc[co][0] = fmaf(wv, patch[ky][kx],         acc[co][0]);
          acc[co][1] = fmaf(wv, patch[ky][kx + 1],     acc[co][1]);
          acc[co][2] = fmaf(wv, patch[ky + 1][kx],     acc[co][2]);
          acc[co][3] = fmaf(wv, patch[ky + 1][kx + 1], acc[co][3]);
        }
  }
#pragma unroll
  for (int co = 0; co < 4; ++co) {
    float r = fmaxf(fmaxf(acc[co][0], acc[co][1]), fmaxf(acc[co][2], acc[co][3]));
    r = fmaxf(r, 0.0f);
    x2h[(((size_t)img * 4 + co) * HC + yo) * WC + xo] = (_Float16)r;
  }
}

// ---------------------------------------------------------------------------
// k_convx: x2h (img,4,64,64) -> xg (img,32,64,64) f16 via WMMA implicit GEMM.
// K = 4 cin * 32 padded taps = 128 (4 K-steps). N = 32 (2 tiles). Block covers
// 2 output rows x 64 cols of one image (8 waves, 16-px M-tiles).
// ---------------------------------------------------------------------------
#define KPX  128
#define KSRX 144            // panel row stride (halves)

__global__ void k_convx(const _Float16* __restrict__ x2h,
                        const float* __restrict__ wx,
                        const float* __restrict__ bx,
                        _Float16* __restrict__ xg) {
  __shared__ _Float16 sA[4 * TR * TW];        // 2176 halves, zero-padded tile
  __shared__ _Float16 swpT[32 * KSRX];        // transposed weight panel
  __shared__ float    sbx[32];

  const int tid = threadIdx.x;
  const int img = blockIdx.x >> 5;            // 8192 blocks: 256 imgs * 32 rowpairs
  const int y0  = (blockIdx.x & 31) << 1;

  // Stage weight panel: swpT[n][k], k = cin*32 + tap (tap >= 25 -> 0).
  for (int i = tid; i < 32 * KPX; i += 256) {
    int n = i >> 7, k = i & 127;
    int cin = k >> 5, tap = k & 31;
    float v = (tap < 25) ? wx[(n * 4 + cin) * 25 + tap] : 0.0f;
    swpT[n * KSRX + k] = (_Float16)v;
  }
  if (tid < 32) sbx[tid] = bx[tid];

  // Stage input tile: rows y0-2 .. y0+3 (halo), rows 6,7 zero, cols -2..65.
  const _Float16* src = x2h + (size_t)img * 4 * HC * WC;
  for (int i = tid; i < 4 * TR * TW; i += 256) {
    int r  = i / TW, rx = i - r * TW;
    int cin = r >> 3, ry = r & 7;
    int gy = y0 - 2 + ry, gx = rx - 2;
    float v = 0.0f;
    if (ry < 6 && gy >= 0 && gy < HC && gx >= 0 && gx < WC)
      v = (float)src[(cin * HC + gy) * WC + gx];
    sA[i] = (_Float16)v;
  }
  __syncthreads();

  const int lane = tid & 31;
  const int wid  = tid >> 5;
  const int y      = y0 + (wid >> 2);
  const int x0     = (wid & 3) << 4;
  const int hi     = lane >> 4;
  const int yloc68 = (y - y0) * TW;
  const int xt     = x0 + (lane & 15);        // tile col (halo-shifted by dx)

  v8f acc0 = {};
  v8f acc1 = {};
  {
    v16h a, b0, b1;
    a  = load_A_tile<0>(sA, yloc68, xt, hi);
    b0 = load_B_panel(swpT, KSRX, 0, 0, lane);
    b1 = load_B_panel(swpT, KSRX, 0, 1, lane);
    acc0 = __builtin_amdgcn_wmma_f32_16x16x32_f16(false, a, false, b0, (short)0, acc0, false, false);
    acc1 = __builtin_amdgcn_wmma_f32_16x16x32_f16(false, a, false, b1, (short)0, acc1, false, false);
    a  = load_A_tile<1>(sA, yloc68, xt, hi);
    b0 = load_B_panel(swpT, KSRX, 1, 0, lane);
    b1 = load_B_panel(swpT, KSRX, 1, 1, lane);
    acc0 = __builtin_amdgcn_wmma_f32_16x16x32_f16(false, a, false, b0, (short)0, acc0, false, false);
    acc1 = __builtin_amdgcn_wmma_f32_16x16x32_f16(false, a, false, b1, (short)0, acc1, false, false);
    a  = load_A_tile<2>(sA, yloc68, xt, hi);
    b0 = load_B_panel(swpT, KSRX, 2, 0, lane);
    b1 = load_B_panel(swpT, KSRX, 2, 1, lane);
    acc0 = __builtin_amdgcn_wmma_f32_16x16x32_f16(false, a, false, b0, (short)0, acc0, false, false);
    acc1 = __builtin_amdgcn_wmma_f32_16x16x32_f16(false, a, false, b1, (short)0, acc1, false, false);
    a  = load_A_tile<3>(sA, yloc68, xt, hi);
    b0 = load_B_panel(swpT, KSRX, 3, 0, lane);
    b1 = load_B_panel(swpT, KSRX, 3, 1, lane);
    acc0 = __builtin_amdgcn_wmma_f32_16x16x32_f16(false, a, false, b0, (short)0, acc0, false, false);
    acc1 = __builtin_amdgcn_wmma_f32_16x16x32_f16(false, a, false, b1, (short)0, acc1, false, false);
  }

  // C/D layout: lanes 0-15: N=lane, M=r; lanes 16-31: N=lane-16, M=r+8.
  const int mbase = hi * 8;
  _Float16* dst = xg + (size_t)img * 32 * (HC * WC);
  {
    int n = lane & 15;
    float bias = sbx[n];
    _Float16* p = dst + (size_t)n * (HC * WC) + y * WC + x0 + mbase;
#pragma unroll
    for (int r = 0; r < 8; ++r) p[r] = (_Float16)(acc0[r] + bias);
  }
  {
    int n = (lane & 15) + 16;
    float bias = sbx[n];
    _Float16* p = dst + (size_t)n * (HC * WC) + y * WC + x0 + mbase;
#pragma unroll
    for (int r = 0; r < 8; ++r) p[r] = (_Float16)(acc1[r] + bias);
  }
}

// ---------------------------------------------------------------------------
// k_lstm: one recurrence step. WMMA conv over h: K = 8 cin * 32 padded taps
// = 256 (8 K-steps). C-frags spilled to LDS, then fused peephole gate math.
// h/c live in d_out (f32); h16 (f16) in ws feeds the next step's A-matrix.
// ---------------------------------------------------------------------------
#define KPH  256
#define KSRH 272            // panel row stride (halves)

__global__ void k_lstm(const _Float16* __restrict__ xg,
                       const float* __restrict__ wh,
                       const float* __restrict__ bh,
                       const float* __restrict__ wci,
                       const float* __restrict__ wcf,
                       const float* __restrict__ wco,
                       float* __restrict__ hout,
                       float* __restrict__ cout,
                       _Float16* __restrict__ h16,
                       int t) {
  __shared__ _Float16 sA[8 * TR * TW];        // 4352 halves, zero-padded h tile
  __shared__ _Float16 swpT[32 * KSRH];        // transposed wh panel
  __shared__ float    sg[8][32 * 16];         // per-wave gate spill, 16 KB

  const int tid  = threadIdx.x;
  const int lane = tid & 31;
  const int wid  = tid >> 5;
  const int b    = blockIdx.x >> 5;           // 1024 blocks: 32 imgs * 32 rowpairs
  const int y0   = (blockIdx.x & 31) << 1;

  if (t > 0) {
    for (int i = tid; i < 32 * KPH; i += 256) {
      int n = i >> 8, k = i & 255;
      int cin = k >> 5, tap = k & 31;
      float v = (tap < 25) ? wh[(n * 8 + cin) * 25 + tap] : 0.0f;
      swpT[n * KSRH + k] = (_Float16)v;
    }
    const _Float16* src = h16 + (size_t)b * 8 * (HC * WC);
    for (int i = tid; i < 8 * TR * TW; i += 256) {
      int r  = i / TW, rx = i - r * TW;
      int cin = r >> 3, ry = r & 7;
      int gy = y0 - 2 + ry, gx = rx - 2;
      float v = 0.0f;
      if (ry < 6 && gy >= 0 && gy < HC && gx >= 0 && gx < WC)
        v = (float)src[(cin * HC + gy) * WC + gx];
      sA[i] = (_Float16)v;
    }
  }
  __syncthreads();

  const int y      = y0 + (wid >> 2);
  const int x0     = (wid & 3) << 4;
  const int hi     = lane >> 4;
  const int yloc68 = (y - y0) * TW;
  const int xt     = x0 + (lane & 15);

  v8f acc0 = {};
  v8f acc1 = {};
  if (t > 0) {
    v16h a, b0, b1;
#define LSTM_KSTEP(KS)                                                         \
    a  = load_A_tile<KS>(sA, yloc68, xt, hi);                                  \
    b0 = load_B_panel(swpT, KSRH, KS, 0, lane);                                \
    b1 = load_B_panel(swpT, KSRH, KS, 1, lane);                                \
    acc0 = __builtin_amdgcn_wmma_f32_16x16x32_f16(false, a, false, b0,         \
                                                  (short)0, acc0, false, false); \
    acc1 = __builtin_amdgcn_wmma_f32_16x16x32_f16(false, a, false, b1,         \
                                                  (short)0, acc1, false, false);
    LSTM_KSTEP(0) LSTM_KSTEP(1) LSTM_KSTEP(2) LSTM_KSTEP(3)
    LSTM_KSTEP(4) LSTM_KSTEP(5) LSTM_KSTEP(6) LSTM_KSTEP(7)
#undef LSTM_KSTEP
  }

  // Spill both C-fragments: sg[wid][n*16 + m] = conv_h gate n at pixel m.
  {
    const int mbase = hi * 8;
    int n0 = lane & 15;
    int n1 = n0 + 16;
#pragma unroll
    for (int r = 0; r < 8; ++r) {
      sg[wid][n0 * 16 + mbase + r] = acc0[r];
      sg[wid][n1 * 16 + mbase + r] = acc1[r];
    }
  }
  __syncthreads();

  // Elementwise gate math: 8 channels x 16 pixels per wave, 4 items per lane.
  const float* sgw = sg[wid];
#pragma unroll
  for (int it = 0; it < 4; ++it) {
    const int ch = (lane >> 4) * 4 + it;        // 0..7
    const int px = lane & 15;
    const int x  = x0 + px;
    const size_t sp    = (size_t)y * WC + x;
    const size_t hcidx = ((size_t)b * 8 + ch) * (HC * WC) + sp;
    const size_t xbase = ((size_t)(t * BB + b) * 32) * (HC * WC) + sp;

    float cprev = (t == 0) ? 0.0f : cout[hcidx];

    float gi = (float)xg[xbase + (size_t)(ch)      * (HC * WC)] + bh[ch]      + sgw[(ch)      * 16 + px];
    float gf = (float)xg[xbase + (size_t)(ch + 8)  * (HC * WC)] + bh[ch + 8]  + sgw[(ch + 8)  * 16 + px];
    float gc = (float)xg[xbase + (size_t)(ch + 16) * (HC * WC)] + bh[ch + 16] + sgw[(ch + 16) * 16 + px];
    float go = (float)xg[xbase + (size_t)(ch + 24) * (HC * WC)] + bh[ch + 24] + sgw[(ch + 24) * 16 + px];

    const size_t pidx = (size_t)ch * (HC * WC) + sp;   // peephole (1,8,64,64)
    float ig = sigmoid_f(gi + cprev * wci[pidx]);
    float fg = sigmoid_f(gf + cprev * wcf[pidx]);
    float cn = fg * cprev + ig * tanh_f(gc);
    float og = sigmoid_f(go + cn * wco[pidx]);
    float hn = og * tanh_f(cn);

    cout[hcidx] = cn;
    hout[hcidx] = hn;
    h16[hcidx]  = (_Float16)hn;
  }
}

// ---------------------------------------------------------------------------
// kernel_launch
// ---------------------------------------------------------------------------
extern "C" void kernel_launch(void* const* d_in, const int* in_sizes, int n_in,
                              void* d_out, int out_size, void* d_ws, size_t ws_size,
                              hipStream_t stream) {
  (void)in_sizes; (void)n_in; (void)out_size; (void)ws_size;

  const float* obs = (const float*)d_in[0];
  const float* w1  = (const float*)d_in[1];
  const float* b1  = (const float*)d_in[2];
  const float* w2  = (const float*)d_in[3];
  const float* b2  = (const float*)d_in[4];
  const float* wx  = (const float*)d_in[5];
  const float* bx  = (const float*)d_in[6];
  const float* wh  = (const float*)d_in[7];
  const float* bh  = (const float*)d_in[8];
  const float* wci = (const float*)d_in[9];
  const float* wcf = (const float*)d_in[10];
  const float* wco = (const float*)d_in[11];

  // Workspace layout (all f16 intermediates; total ~106 MB, L2-resident):
  //   x1h : 256*4*128*128 f16 = 32 MB
  //   x2h : 256*4*64*64   f16 =  8 MB
  //   xg  : 256*32*64*64  f16 = 64 MB
  //   h16 : 32*8*64*64    f16 =  2 MB
  char* ws = (char*)d_ws;
  _Float16* x1h = (_Float16*)ws;
  _Float16* x2h = (_Float16*)(ws + (size_t)33554432);
  _Float16* xg  = (_Float16*)(ws + (size_t)33554432 + 8388608);
  _Float16* h16 = (_Float16*)(ws + (size_t)33554432 + 8388608 + 67108864);

  float* hout = (float*)d_out;                    // (32,8,64,64)
  float* cout = hout + (size_t)BB * 8 * HC * WC;  // (32,8,64,64)

  k_conv1<<<16384, 256, 0, stream>>>(obs, w1, b1, x1h);
  k_conv2<<<4096,  256, 0, stream>>>(x1h, w2, b2, x2h);
  k_convx<<<8192,  256, 0, stream>>>(x2h, wx, bx, xg);
  for (int t = 0; t < TT; ++t)
    k_lstm<<<1024, 256, 0, stream>>>(xg, wh, bh, wci, wcf, wco,
                                     hout, cout, h16, t);
}
